// CrossAttention_51290499449411
// MI455X (gfx1250) — matrix operand order repaired and checked by reading, not compiled, not run on hardware
//
#include <hip/hip_runtime.h>
#include <hip/hip_bf16.h>

// ---------------------------------------------------------------------------
// CDNA5 (gfx1250, wave32) dual cross-attention block.
// All GEMMs: v_wmma_f32_16x16x32_bf16, f32 accumulate.
// LDS operands are stored K-contiguous per fragment row so every fragment is
// two aligned ds_load_b128; staging uses global_load_b128 + packed bf16 stores.
// ---------------------------------------------------------------------------

typedef __attribute__((ext_vector_type(16))) __bf16 bf16x16;
typedef __attribute__((ext_vector_type(8)))  __bf16 bf16x8;
typedef __attribute__((ext_vector_type(4)))  __bf16 bf16x4;
typedef __attribute__((ext_vector_type(8)))  float  f32x8;

#define BATCH   4
#define CDIM    192
#define C3      576
#define HDIM    128
#define WDIM    128
#define NSP     (HDIM * WDIM)          // 16384
#define NHEADS  4
#define HC      48                     // channels per head
#define KCHUNK  32                     // K-split factor for the logits GEMM

__device__ __forceinline__ f32x8 wmma_bf16(bf16x16 a, bf16x16 b, f32x8 c) {
    return __builtin_amdgcn_wmma_f32_16x16x32_bf16(false, a, false, b,
                                                   (short)0, c, false, false);
}

// Fragment = two aligned 16B LDS vector loads (ds_load_b128).
__device__ __forceinline__ bf16x16 ld_frag(const __bf16* lo, const __bf16* hi) {
    bf16x8 a = *(const bf16x8*)lo;
    bf16x8 b = *(const bf16x8*)hi;
    bf16x16 r;
    #pragma unroll
    for (int i = 0; i < 8; ++i) { r[i] = a[i]; r[8 + i] = b[i]; }
    return r;
}

__device__ __forceinline__ bf16x4 cvt4(float4 v) {
    bf16x4 p = { (__bf16)v.x, (__bf16)v.y, (__bf16)v.z, (__bf16)v.w };
    return p;
}
__device__ __forceinline__ bf16x4 cvt4s(float4 v, float s) {
    bf16x4 p = { (__bf16)(v.x * s), (__bf16)(v.y * s),
                 (__bf16)(v.z * s), (__bf16)(v.w * s) };
    return p;
}

// ===========================================================================
// Kernel 1: 1x1 conv as GEMM  OUT[b,o,n] = sum_i W[o,i] * IN[b,i,n]
// Block tile M=64 x N=64, K steps of 32. 8 waves: wave w -> M tile (w>>1),
// two N tiles at (w&1)*32 and +16 (A fragment reused for both WMMAs).
// ===========================================================================
__global__ void __launch_bounds__(256)
conv1x1_wmma(const float* __restrict__ in,   // [B, Ci, NSP]
             const float* __restrict__ w,    // [Co, Ci]
             float* __restrict__ out,        // [B, Co, NSP]
             int Ci, int Co)
{
    __shared__ alignas(16) __bf16 Asm[64][40];   // [M][K] row-major, stride 80B
    __shared__ alignas(16) __bf16 BsT[64][40];   // [N][K] (K-contiguous per col)

    const int b   = blockIdx.z;
    const int o0  = blockIdx.y * 64;
    const int n0  = blockIdx.x * 64;
    const int tid = threadIdx.x;
    const int lane = tid & 31;
    const int wid  = tid >> 5;
    const int mw = wid >> 1;          // 0..3
    const int nb = (wid & 1) * 32;    // 0 or 32

    const float* inb = in + (size_t)b * Ci * NSP;

    f32x8 acc0 = {}, acc1 = {};

    for (int k0 = 0; k0 < Ci; k0 += 32) {
        // --- stage A: 64x32 weights, one float4 per thread x2 ---
        #pragma unroll
        for (int t = 0; t < 2; ++t) {
            int i = tid + t * 256;
            int r = i >> 3, c4 = (i & 7) << 2;
            float4 v = *(const float4*)&w[(size_t)(o0 + r) * Ci + k0 + c4];
            *(bf16x4*)&Asm[r][c4] = cvt4(v);
        }
        // --- stage B: 32x64 activations, transposed into [N][K] ---
        #pragma unroll
        for (int t = 0; t < 2; ++t) {
            int i = tid + t * 256;
            int r = i >> 4, c4 = (i & 15) << 2;   // r = K row, c4 = N col
            float4 v = *(const float4*)&inb[(size_t)(k0 + r) * NSP + n0 + c4];
            BsT[c4 + 0][r] = (__bf16)v.x;
            BsT[c4 + 1][r] = (__bf16)v.y;
            BsT[c4 + 2][r] = (__bf16)v.z;
            BsT[c4 + 3][r] = (__bf16)v.w;
        }
        if (k0 + 32 < Ci)   // gfx1250 global_prefetch_b8 on next K slab
            __builtin_prefetch(&inb[(size_t)(k0 + 32 + (tid >> 3)) * NSP + n0 + ((tid & 7) << 3)], 0, 1);
        __syncthreads();

        const int row = (lane & 15) + mw * 16;
        const int kb  = (lane >> 4) * 8;
        bf16x16 af = ld_frag(&Asm[row][kb], &Asm[row][kb + 16]);

        const int kb2 = (lane >> 4) * 8;
        const int c0  = nb + (lane & 15);
        bf16x16 bf0 = ld_frag(&BsT[c0][kb2],      &BsT[c0][kb2 + 16]);
        bf16x16 bf1 = ld_frag(&BsT[c0 + 16][kb2], &BsT[c0 + 16][kb2 + 16]);

        acc0 = wmma_bf16(af, bf0, acc0);
        acc1 = wmma_bf16(af, bf1, acc1);
        __syncthreads();
    }

    float* outb = out + (size_t)b * Co * NSP;
    const int colg  = n0 + nb + (lane & 15);
    const int rbase = o0 + mw * 16 + ((lane >> 4) ? 8 : 0);
    #pragma unroll
    for (int r = 0; r < 8; ++r) {
        outb[(size_t)(rbase + r) * NSP + colg]      = acc0[r];
        outb[(size_t)(rbase + r) * NSP + colg + 16] = acc1[r];
    }
}

// ===========================================================================
// Kernel 2: depthwise 3x3 conv, zero padding, fp32 (bandwidth bound)
// ===========================================================================
__global__ void __launch_bounds__(256)
dwconv3x3(const float* __restrict__ in,   // [B*C3, H, W]
          const float* __restrict__ w,    // [C3, 9]
          float* __restrict__ out,
          long total)
{
    long idx = (long)blockIdx.x * blockDim.x + threadIdx.x;
    if (idx >= total) return;
    int  xx = (int)(idx % WDIM);
    long t  = idx / WDIM;
    int  yy = (int)(t % HDIM);
    long bc = t / HDIM;
    const float* wp = w + (bc % C3) * 9;
    const float* p  = in + (bc * HDIM + yy) * (long)WDIM + xx;

    float s = 0.f;
    #pragma unroll
    for (int dy = -1; dy <= 1; ++dy) {
        int y = yy + dy;
        if (y < 0 || y >= HDIM) continue;
        #pragma unroll
        for (int dx = -1; dx <= 1; ++dx) {
            int x = xx + dx;
            if (x < 0 || x >= WDIM) continue;
            s += wp[(dy + 1) * 3 + (dx + 1)] * p[dy * WDIM + dx];
        }
    }
    out[idx] = s;
}

// ===========================================================================
// Kernel 3: per-row L2 norm -> scale = 1 / max(||row||, eps), float4 loads
// ===========================================================================
__global__ void __launch_bounds__(256)
rownorm(const float* __restrict__ src, float* __restrict__ scale, long bstride)
{
    const int r = blockIdx.x;            // [0, B*CDIM)
    const int b = r / CDIM, c = r % CDIM;
    const float4* p4 = (const float4*)(src + (size_t)b * bstride + (size_t)c * NSP);

    float s = 0.f;
    for (int i = threadIdx.x; i < NSP / 4; i += 256) {
        float4 v = p4[i];
        s += v.x * v.x + v.y * v.y + v.z * v.z + v.w * v.w;
    }
    #pragma unroll
    for (int off = 16; off > 0; off >>= 1)
        s += __shfl_down(s, off, 32);

    __shared__ float red[8];
    if ((threadIdx.x & 31) == 0) red[threadIdx.x >> 5] = s;
    __syncthreads();
    if (threadIdx.x == 0) {
        float tot = 0.f;
        #pragma unroll
        for (int i = 0; i < 8; ++i) tot += red[i];
        scale[r] = 1.0f / fmaxf(sqrtf(tot), 1e-12f);
    }
}

// ===========================================================================
// Kernel 4: attention logit partials. Grid (KCHUNK, B*NHEADS); each block
// reduces a 512-column slab of K into a 48x48 partial. 9 waves, one 16x16
// tile each. Normalization fused at LDS load.
// ===========================================================================
__global__ void __launch_bounds__(288)
attn_logits_part(const float* __restrict__ qbase,
                 const float* __restrict__ kbase,
                 const float* __restrict__ qscale,
                 const float* __restrict__ kscale,
                 float* __restrict__ part,    // [B*NHEADS, KCHUNK, 48, 48]
                 long bstride)
{
    __shared__ alignas(16) __bf16 Qs[HC][40];   // [c][k]
    __shared__ alignas(16) __bf16 Ks[HC][40];   // [d][k] (B operand, K-contig)

    const int chunk = blockIdx.x;
    const int bh    = blockIdx.y;
    const int b  = bh / NHEADS, h = bh % NHEADS;
    const int tid = threadIdx.x;
    const int lane = tid & 31;
    const int wid  = tid >> 5;       // 0..8
    const int mt = wid / 3;
    const int nt = wid % 3;

    const float* qp = qbase + (size_t)b * bstride + (size_t)(h * HC) * NSP;
    const float* kp = kbase + (size_t)b * bstride + (size_t)(h * HC) * NSP;
    const float* qs = qscale + b * CDIM + h * HC;
    const float* ks = kscale + b * CDIM + h * HC;

    const int nbeg = chunk * (NSP / KCHUNK);
    const int nend = nbeg + (NSP / KCHUNK);

    f32x8 acc = {};

    for (int n0 = nbeg; n0 < nend; n0 += 32) {
        // 384 float4 for Q + 384 for K, over 288 threads
        for (int i = tid; i < 768; i += 288) {
            int j  = (i < 384) ? i : i - 384;
            int r  = j >> 3, c4 = (j & 7) << 2;
            if (i < 384) {
                float4 v = *(const float4*)&qp[(size_t)r * NSP + n0 + c4];
                *(bf16x4*)&Qs[r][c4] = cvt4s(v, qs[r]);
            } else {
                float4 v = *(const float4*)&kp[(size_t)r * NSP + n0 + c4];
                *(bf16x4*)&Ks[r][c4] = cvt4s(v, ks[r]);
            }
        }
        __syncthreads();

        const int row = (lane & 15) + mt * 16;
        const int kb  = (lane >> 4) * 8;
        bf16x16 af = ld_frag(&Qs[row][kb], &Qs[row][kb + 16]);

        const int dcol = (lane & 15) + nt * 16;
        const int kb2  = (lane >> 4) * 8;
        bf16x16 bfr = ld_frag(&Ks[dcol][kb2], &Ks[dcol][kb2 + 16]);

        acc = wmma_bf16(af, bfr, acc);
        __syncthreads();
    }

    float* pp = part + ((size_t)bh * KCHUNK + chunk) * (HC * HC);
    const int col   = nt * 16 + (lane & 15);
    const int rbase = mt * 16 + ((lane >> 4) ? 8 : 0);
    #pragma unroll
    for (int r = 0; r < 8; ++r)
        pp[(size_t)(rbase + r) * HC + col] = acc[r];
}

// ===========================================================================
// Kernel 5: deterministic partial reduction + temperature scale
// ===========================================================================
__global__ void __launch_bounds__(256)
attn_reduce(const float* __restrict__ part, const float* __restrict__ temp,
            float* __restrict__ attn)       // [B*NHEADS, 48, 48]
{
    const int bh = blockIdx.x;
    const float t = temp[bh % NHEADS];
    const float* p = part + (size_t)bh * KCHUNK * (HC * HC);
    for (int e = threadIdx.x; e < HC * HC; e += 256) {
        float s = 0.f;
        #pragma unroll 8
        for (int ch = 0; ch < KCHUNK; ++ch)
            s += p[(size_t)ch * (HC * HC) + e];
        attn[(size_t)bh * (HC * HC) + e] = s * t;
    }
}

// ===========================================================================
// Kernel 6: softmax over last dim (48), in place
// ===========================================================================
__global__ void __launch_bounds__(64)
softmax48(float* __restrict__ attn)
{
    const int bh = blockIdx.x;
    const int t  = threadIdx.x;
    if (t < HC) {
        float* row = attn + ((size_t)bh * HC + t) * HC;
        float m = row[0];
        for (int j = 1; j < HC; ++j) m = fmaxf(m, row[j]);
        float s = 0.f;
        for (int j = 0; j < HC; ++j) { float e = __expf(row[j] - m); row[j] = e; s += e; }
        float inv = 1.0f / s;
        for (int j = 0; j < HC; ++j) row[j] *= inv;
    }
}

// ===========================================================================
// Kernel 7: OUT[b,h,c,n] = sum_d attn[c,d] * v[d,n]   (K=48 padded to 64)
// Block = (b,h) x 128-column slab. 8 waves; wave owns a 16-col slice, loops
// the 3 M tiles; 2 K steps -> 6 WMMAs per wave.
// ===========================================================================
__global__ void __launch_bounds__(256)
attn_apply_wmma(const float* __restrict__ attn,  // [B*NHEADS, 48, 48]
                const float* __restrict__ vbase,
                float* __restrict__ out,         // [B, CDIM, NSP]
                long bstride)
{
    __shared__ alignas(16) __bf16 Aw[HC][72];    // [c][d], d padded 48->64
    __shared__ alignas(16) __bf16 VsT[128][72];  // [n][d], d padded 48->64

    const int n0 = blockIdx.x * 128;
    const int bh = blockIdx.y;
    const int b  = bh / NHEADS, h = bh % NHEADS;
    const int tid = threadIdx.x;
    const int lane = tid & 31;
    const int wid  = tid >> 5;

    const float* ap = attn + (size_t)bh * (HC * HC);
    const float* vp = vbase + (size_t)b * bstride + (size_t)(h * HC) * NSP;

    // attn tile (48x48) -> Aw
    for (int i = tid; i < 576; i += 256) {
        int r = i / 12, c4 = (i % 12) << 2;
        float4 v = *(const float4*)&ap[(size_t)r * HC + c4];
        *(bf16x4*)&Aw[r][c4] = cvt4(v);
    }
    // zero-pad d = 48..63 of Aw
    for (int i = tid; i < 192; i += 256) {
        int r = i >> 2, c4 = 48 + ((i & 3) << 2);
        bf16x4 z = {};
        *(bf16x4*)&Aw[r][c4] = z;
    }
    // V tile (48x128) -> VsT transposed
    for (int i = tid; i < 1536; i += 256) {
        int r = i >> 5, c4 = (i & 31) << 2;      // r = d, c4 = n
        float4 v = *(const float4*)&vp[(size_t)r * NSP + n0 + c4];
        VsT[c4 + 0][r] = (__bf16)v.x;
        VsT[c4 + 1][r] = (__bf16)v.y;
        VsT[c4 + 2][r] = (__bf16)v.z;
        VsT[c4 + 3][r] = (__bf16)v.w;
    }
    // zero-pad d = 48..63 of VsT
    for (int i = tid; i < 512; i += 256) {
        int n = i >> 2, c4 = 48 + ((i & 3) << 2);
        bf16x4 z = {};
        *(bf16x4*)&VsT[n][c4] = z;
    }
    __syncthreads();

    const int col = wid * 16 + (lane & 15);
    float* ob = out + (size_t)b * CDIM * NSP + (size_t)(h * HC) * NSP;

    #pragma unroll
    for (int mt = 0; mt < 3; ++mt) {
        f32x8 acc = {};
        #pragma unroll
        for (int ks = 0; ks < 64; ks += 32) {
            const int row = (lane & 15) + mt * 16;
            const int kb  = ks + (lane >> 4) * 8;
            bf16x16 af = ld_frag(&Aw[row][kb], &Aw[row][kb + 16]);
            const int kb2 = ks + (lane >> 4) * 8;
            bf16x16 bfr = ld_frag(&VsT[col][kb2], &VsT[col][kb2 + 16]);
            acc = wmma_bf16(af, bfr, acc);
        }
        const int rbase = mt * 16 + ((lane >> 4) ? 8 : 0);
        #pragma unroll
        for (int r = 0; r < 8; ++r)
            ob[(size_t)(rbase + r) * NSP + (n0 + col)] = acc[r];
    }
}

// ===========================================================================
// Host-side orchestration
// ===========================================================================
extern "C" void kernel_launch(void* const* d_in, const int* in_sizes, int n_in,
                              void* d_out, int out_size, void* d_ws, size_t ws_size,
                              hipStream_t stream)
{
    const float* x         = (const float*)d_in[0];
    const float* ctx       = (const float*)d_in[1];
    const float* x_qkv_w   = (const float*)d_in[2];
    const float* x_dw_w    = (const float*)d_in[3];
    const float* x_po_w    = (const float*)d_in[4];
    const float* ctx_qkv_w = (const float*)d_in[5];
    const float* ctx_dw_w  = (const float*)d_in[6];
    const float* ctx_po_w  = (const float*)d_in[7];
    const float* temp      = (const float*)d_in[8];

    const size_t seg  = (size_t)BATCH * C3 * NSP;     // floats per big segment
    const size_t aseg = (size_t)BATCH * CDIM * NSP;
    float* ws = (float*)d_ws;

    float* x_pre    = ws;             // later reused as x_att / ctx_att
    float* x_qkv    = ws + seg;
    float* ctx_pre  = ws + 2 * seg;   // later reused for scales/attn/partials
    float* ctx_qkv  = ws + 3 * seg;

    float* x_att    = x_pre;
    float* ctx_att  = x_pre + aseg;

    float* sc_xq  = ctx_pre;
    float* sc_xk  = ctx_pre + 1024;
    float* sc_cq  = ctx_pre + 2048;
    float* sc_ck  = ctx_pre + 3072;
    float* attn1  = ctx_pre + 4096;
    float* attn2  = attn1 + (size_t)BATCH * NHEADS * HC * HC;
    float* part1  = attn2 + (size_t)BATCH * NHEADS * HC * HC;
    float* part2  = part1 + (size_t)BATCH * NHEADS * KCHUNK * HC * HC;

    const long bstr = (long)C3 * NSP;
    const long total_dw = (long)BATCH * C3 * NSP;

    // --- 1) qkv 1x1 convs + depthwise 3x3 ---
    {
        dim3 g(NSP / 64, C3 / 64, BATCH);
        int nb = (int)((total_dw + 255) / 256);
        conv1x1_wmma<<<g, 256, 0, stream>>>(x,   x_qkv_w,   x_pre,   CDIM, C3);
        dwconv3x3<<<nb, 256, 0, stream>>>(x_pre, x_dw_w, x_qkv, total_dw);
        conv1x1_wmma<<<g, 256, 0, stream>>>(ctx, ctx_qkv_w, ctx_pre, CDIM, C3);
        dwconv3x3<<<nb, 256, 0, stream>>>(ctx_pre, ctx_dw_w, ctx_qkv, total_dw);
    }

    // --- 2) row L2-norm scales ---
    {
        int rows = BATCH * CDIM;
        rownorm<<<rows, 256, 0, stream>>>(x_qkv,                        sc_xq, bstr);
        rownorm<<<rows, 256, 0, stream>>>(x_qkv   + (size_t)CDIM * NSP, sc_xk, bstr);
        rownorm<<<rows, 256, 0, stream>>>(ctx_qkv,                      sc_cq, bstr);
        rownorm<<<rows, 256, 0, stream>>>(ctx_qkv + (size_t)CDIM * NSP, sc_ck, bstr);
    }

    // --- 3) attention logits (K-split WMMA) + reduce + softmax ---
    {
        dim3 g(KCHUNK, BATCH * NHEADS);
        attn_logits_part<<<g, 288, 0, stream>>>(
            x_qkv, ctx_qkv + (size_t)CDIM * NSP, sc_xq, sc_ck, part1, bstr);
        attn_logits_part<<<g, 288, 0, stream>>>(
            ctx_qkv, x_qkv + (size_t)CDIM * NSP, sc_cq, sc_xk, part2, bstr);
        attn_reduce<<<BATCH * NHEADS, 256, 0, stream>>>(part1, temp, attn1);
        attn_reduce<<<BATCH * NHEADS, 256, 0, stream>>>(part2, temp, attn2);
        softmax48<<<BATCH * NHEADS, 64, 0, stream>>>(attn1);
        softmax48<<<BATCH * NHEADS, 64, 0, stream>>>(attn2);
    }

    // --- 4) attn @ V (WMMA) ---
    {
        dim3 g(NSP / 128, BATCH * NHEADS);
        attn_apply_wmma<<<g, 256, 0, stream>>>(attn1, ctx_qkv + (size_t)2 * CDIM * NSP,
                                               ctx_att, bstr);
        attn_apply_wmma<<<g, 256, 0, stream>>>(attn2, x_qkv   + (size_t)2 * CDIM * NSP,
                                               x_att,   bstr);
    }

    // --- 5) output projections into d_out (x_out first, then ctx_out) ---
    {
        float* out_x   = (float*)d_out;
        float* out_ctx = (float*)d_out + aseg;
        dim3 g(NSP / 64, CDIM / 64, BATCH);
        conv1x1_wmma<<<g, 256, 0, stream>>>(x_att,   x_po_w,   out_x,   CDIM, CDIM);
        conv1x1_wmma<<<g, 256, 0, stream>>>(ctx_att, ctx_po_w, out_ctx, CDIM, CDIM);
    }
}
